// GRUModel_64321430225551
// MI455X (gfx1250) — matrix operand order repaired and checked by reading, not compiled, not run on hardware
//
#include <hip/hip_runtime.h>
#include <hip/hip_bf16.h>

// ---------------------------------------------------------------------------
// GRU LM forward on MI455X (gfx1250): bf16 WMMA (v_wmma_f32_16x16x32_bf16)
// for all GEMMs, f32 accumulate, f32 gate math.
// Recurrence: 16 persistent blocks, h staged in LDS (ds_load_b128 -> wmma),
// Whh streamed exactly once per step from L2, device barrier via L2 atomics.
// ---------------------------------------------------------------------------

typedef __attribute__((ext_vector_type(16))) __bf16 v16bf;
typedef __attribute__((ext_vector_type(8)))  __bf16 v8bf;
typedef __attribute__((ext_vector_type(8)))  float  v8f;

#define GRU_V 128
#define GRU_E 512
#define GRU_H 1024
#define GRU_B 64
#define GRU_T 512

// ---- WMMA fragment loaders (ISA 7.12.2 layouts, wave32) --------------------
// A (16x32 bf16, MxK): lane l<16 -> row M=l, K = {k0..k0+7, k0+16..k0+23}
//                      lane l>=16 -> row M=l-16, K = {k0+8..15, k0+24..31}
__device__ __forceinline__ v16bf load_a_frag(const __bf16* __restrict__ A,
                                             int lda, int row0, int k0, int lane) {
  int m  = row0 + (lane & 15);
  int hs = lane >> 4;
  const __bf16* p = A + (size_t)m * lda + k0 + (hs << 3);
  union { v16bf v; v8bf h[2]; } u;
  u.h[0] = *(const v8bf*)(p);
  u.h[1] = *(const v8bf*)(p + 16);
  return u.v;
}

__device__ __forceinline__ v16bf load_a_frag_lds(const __bf16* A,
                                                 int lda, int row0, int k0, int lane) {
  int m  = row0 + (lane & 15);
  int hs = lane >> 4;
  const __bf16* p = A + (size_t)m * lda + k0 + (hs << 3);
  union { v16bf v; v8bf h[2]; } u;
  u.h[0] = *(const v8bf*)(p);       // ds_load_b128
  u.h[1] = *(const v8bf*)(p + 16);  // ds_load_b128
  return u.v;
}

// B (32x16 bf16, KxN) built from row-major weights W[N,K] (computes A @ W^T):
// lane l<16 -> column n=l, K = {k0..k0+7, k0+16..k0+23} ; lane l>=16 -> col n=l-16, K = {k0+8..15, k0+24..31}
__device__ __forceinline__ v16bf load_b_frag(const __bf16* __restrict__ W,
                                             int ldw, int col0, int k0, int lane) {
  int n  = col0 + (lane & 15);
  int hs = lane >> 4;
  const __bf16* p = W + (size_t)n * ldw + k0 + (hs << 3);
  union { v16bf v; v8bf h[2]; } u;
  u.h[0] = *(const v8bf*)(p);
  u.h[1] = *(const v8bf*)(p + 16);
  return u.v;
}

// ---- prep kernels ----------------------------------------------------------
__global__ void f32_to_bf16_kernel(const float* __restrict__ src,
                                   __bf16* __restrict__ dst, int n) {
  int i = blockIdx.x * blockDim.x + threadIdx.x;
  if (i < n) dst[i] = (__bf16)src[i];
}

__global__ void gather_emb_kernel(const int* __restrict__ x,
                                  const float* __restrict__ emb,
                                  __bf16* __restrict__ A0, int total) {
  int i = blockIdx.x * blockDim.x + threadIdx.x;
  if (i < total) {
    int row = i >> 9;            // / E (=512)
    int col = i & (GRU_E - 1);
    int tok = x[row];
    A0[i] = (__bf16)emb[(size_t)tok * GRU_E + col];
  }
}

__global__ void zero_state_kernel(__bf16* __restrict__ hbuf, int n,
                                  unsigned* __restrict__ bar) {
  int i = blockIdx.x * blockDim.x + threadIdx.x;
  if (i < n) hbuf[i] = (__bf16)0.0f;
  if (i == 0) { bar[0] = 0u; bar[1] = 0u; }
}

// ---- generic bf16 WMMA GEMM: C[M,N] = A[M,K] @ W[N,K]^T + bias -------------
// One wave computes a 16x64 output strip (A fragment reused across 4 B tiles).
// remapTB: write logical row r=b*T+t to output row t*B+b (gx stored [T,B,3H]).
__global__ void wmma_gemm_bias_kernel(const __bf16* __restrict__ A, int lda,
                                      const __bf16* __restrict__ W, int ldw,
                                      const float* __restrict__ bias,
                                      float* __restrict__ C, int ldc,
                                      int M, int N, int K,
                                      int Tn, int Bn, int remapTB) {
  int lane = threadIdx.x & 31;
  int wave = threadIdx.x >> 5;
  int ng4  = N >> 6;  // groups of 64 columns
  long gid = (long)blockIdx.x * (blockDim.x >> 5) + wave;
  long total = (long)(M >> 4) * ng4;
  if (gid >= total) return;
  int mt   = (int)(gid / ng4);
  int ng   = (int)(gid % ng4);
  int row0 = mt << 4;
  int col0 = ng << 6;

  v8f acc0 = {}, acc1 = {}, acc2 = {}, acc3 = {};
  for (int k0 = 0; k0 < K; k0 += 32) {
    v16bf a  = load_a_frag(A, lda, row0, k0, lane);
    v16bf b0 = load_b_frag(W, ldw, col0 +  0, k0, lane);
    v16bf b1 = load_b_frag(W, ldw, col0 + 16, k0, lane);
    v16bf b2 = load_b_frag(W, ldw, col0 + 32, k0, lane);
    v16bf b3 = load_b_frag(W, ldw, col0 + 48, k0, lane);
    acc0 = __builtin_amdgcn_wmma_f32_16x16x32_bf16(false, a, false, b0, (short)0, acc0, false, false);
    acc1 = __builtin_amdgcn_wmma_f32_16x16x32_bf16(false, a, false, b1, (short)0, acc1, false, false);
    acc2 = __builtin_amdgcn_wmma_f32_16x16x32_bf16(false, a, false, b2, (short)0, acc2, false, false);
    acc3 = __builtin_amdgcn_wmma_f32_16x16x32_bf16(false, a, false, b3, (short)0, acc3, false, false);
  }

  int cl = lane & 15, hs = lane >> 4;
  float bi0 = bias[col0 +  0 + cl];
  float bi1 = bias[col0 + 16 + cl];
  float bi2 = bias[col0 + 32 + cl];
  float bi3 = bias[col0 + 48 + cl];
#pragma unroll
  for (int r = 0; r < 8; ++r) {
    int row = row0 + r + (hs << 3);
    size_t orow;
    if (remapTB) {
      int b_ = row / Tn, t_ = row % Tn;
      orow = (size_t)t_ * Bn + b_;
    } else {
      orow = (size_t)row;
    }
    float* cp = C + orow * (size_t)ldc + col0 + cl;
    cp[ 0] = acc0[r] + bi0;
    cp[16] = acc1[r] + bi1;
    cp[32] = acc2[r] + bi2;
    cp[48] = acc3[r] + bi3;
  }
}

// ---- persistent GRU recurrence ---------------------------------------------
// Grid: 16 blocks x 512 threads (16 waves). Block owns 64 hidden columns.
// Per step:
//   1) stage h[64,1024] bf16 into LDS cooperatively (128 KB)
//   2) waves 0..11 = (gate r/z/n) x (4 col-tiles): 4 independent wmma chains
//      (4 M-tiles sharing one Whh B-fragment), A fed by ds_load_b128
//   3) all 512 threads combine gates (f32), h state held in registers
//   4) device barrier via monotonic L2 atomics (s_sleep spin)
// Dynamic LDS: 64*1024*2 (h stage) + 3*64*64*4 (gate tiles) = 180224 B.
__global__ void __launch_bounds__(512)
gru_scan_kernel(const __bf16* __restrict__ Whh,   // [3H,H] bf16
                const float*  __restrict__ bhh,   // [3H]
                const float*  __restrict__ gx,    // [T,B,3H] f32
                __bf16* __restrict__ hbuf,        // [2,B,H] bf16 (zeroed)
                __bf16* __restrict__ obuf,        // [B*T,H] bf16, row b*T+t
                float*  __restrict__ hfinal,      // [B,H] f32
                int Tlen, unsigned* __restrict__ bar, int nblocks) {
  const int H = GRU_H, B = GRU_B;
  extern __shared__ __align__(16) char smem[];
  __bf16* lds_h = (__bf16*)smem;                    // [64][1024] bf16
  float*  gh    = (float*)(smem + (size_t)B * H * 2); // [3][64][64] f32

  int tid   = threadIdx.x;
  int lane  = tid & 31;
  int wave  = tid >> 5;           // 0..15
  int gate  = wave >> 2;          // 0..3 (3 = idle in GEMM phase)
  int ntile = wave & 3;           // col tile within block's 64 cols
  int hc0   = blockIdx.x << 6;    // 64 hidden columns per block
  int cl = lane & 15, hs = lane >> 4;
  int coll = (ntile << 4) + cl;   // local column for GEMM waves
  int wr0  = gate * H + hc0 + (ntile << 4);  // Whh row base for this wave

  float bb = (gate < 3) ? bhh[wr0 + cl] : 0.0f;

  // combine-phase mapping: 512 threads = 8 row-slots x 64 cols
  int c_col  = tid & 63;          // local column 0..63
  int c_rowb = tid >> 6;          // row base 0..7 (rows c_rowb + 8*rr)
  int jglob  = hc0 + c_col;
  float hreg[8];
#pragma unroll
  for (int rr = 0; rr < 8; ++rr) hreg[rr] = 0.0f;

  for (int t = 0; t < Tlen; ++t) {
    const __bf16* hread  = hbuf + (size_t)(t & 1) * B * H;
    __bf16*       hwrite = hbuf + (size_t)((t + 1) & 1) * B * H;

    // ---- 1) stage h into LDS (16 x b128 per thread) ----
    {
      const v8bf* src = (const v8bf*)hread;
      v8bf*       dst = (v8bf*)lds_h;
#pragma unroll
      for (int i = 0; i < (GRU_B * GRU_H / 8) / 512; ++i)
        dst[tid + i * 512] = src[tid + i * 512];
    }
    __syncthreads();

    // ---- 2) gh = h @ Whh^T for this wave's 64x16 strip ----
    if (gate < 3) {
      v8f acc0 = {}, acc1 = {}, acc2 = {}, acc3 = {};
      for (int k0 = 0; k0 < H; k0 += 32) {
        v16bf bf = load_b_frag(Whh, H, wr0, k0, lane);       // global (L2)
        v16bf a0 = load_a_frag_lds(lds_h, H,  0, k0, lane);  // LDS
        v16bf a1 = load_a_frag_lds(lds_h, H, 16, k0, lane);
        v16bf a2 = load_a_frag_lds(lds_h, H, 32, k0, lane);
        v16bf a3 = load_a_frag_lds(lds_h, H, 48, k0, lane);
        acc0 = __builtin_amdgcn_wmma_f32_16x16x32_bf16(false, a0, false, bf, (short)0, acc0, false, false);
        acc1 = __builtin_amdgcn_wmma_f32_16x16x32_bf16(false, a1, false, bf, (short)0, acc1, false, false);
        acc2 = __builtin_amdgcn_wmma_f32_16x16x32_bf16(false, a2, false, bf, (short)0, acc2, false, false);
        acc3 = __builtin_amdgcn_wmma_f32_16x16x32_bf16(false, a3, false, bf, (short)0, acc3, false, false);
      }
      float* ghg = gh + (size_t)gate * (GRU_B * 64);
#pragma unroll
      for (int r = 0; r < 8; ++r) {
        int rb = r + (hs << 3);
        ghg[( 0 + rb) * 64 + coll] = acc0[r] + bb;
        ghg[(16 + rb) * 64 + coll] = acc1[r] + bb;
        ghg[(32 + rb) * 64 + coll] = acc2[r] + bb;
        ghg[(48 + rb) * 64 + coll] = acc3[r] + bb;
      }
    }
    __syncthreads();

    // ---- 3) gate combine: 8 elements per thread, h in registers ----
    {
      const float* gxt = gx + (size_t)t * (B * 3 * GRU_H);
#pragma unroll
      for (int rr = 0; rr < 8; ++rr) {
        int row = c_rowb + (rr << 3);     // batch row 0..63
        const float* gxr = gxt + (size_t)row * (3 * GRU_H);
        float ghr = gh[(0 * GRU_B + row) * 64 + c_col];
        float ghz = gh[(1 * GRU_B + row) * 64 + c_col];
        float ghn = gh[(2 * GRU_B + row) * 64 + c_col];
        float rg  = 1.0f / (1.0f + __expf(-(gxr[jglob] + ghr)));
        float zg  = 1.0f / (1.0f + __expf(-(gxr[GRU_H + jglob] + ghz)));
        float ngt = tanhf(gxr[2 * GRU_H + jglob] + rg * ghn);
        float hn  = (1.0f - zg) * ngt + zg * hreg[rr];
        hreg[rr]  = hn;
        __bf16 hb = (__bf16)hn;
        hwrite[(size_t)row * H + jglob] = hb;
        obuf[((size_t)row * Tlen + t) * H + jglob] = hb;
        if (t == Tlen - 1) hfinal[(size_t)row * H + jglob] = hn;
      }
    }
    __syncthreads();
    __threadfence();   // make hwrite visible device-wide before the barrier

    // ---- 4) device-wide arrive/spin barrier (monotonic, no reset) ----
    if (tid == 0) {
      unsigned arrived = atomicAdd(&bar[0], 1u) + 1u;
      unsigned target  = (unsigned)nblocks * (unsigned)(t + 1);
      if (arrived == target) {
        atomicAdd(&bar[1], 1u);
      } else {
        while (atomicAdd(&bar[1], 0u) < (unsigned)(t + 1))
          __builtin_amdgcn_s_sleep(2);
      }
    }
    __syncthreads();
  }
}

// ---------------------------------------------------------------------------
extern "C" void kernel_launch(void* const* d_in, const int* in_sizes, int n_in,
                              void* d_out, int out_size, void* d_ws, size_t ws_size,
                              hipStream_t stream) {
  (void)in_sizes; (void)n_in; (void)out_size; (void)ws_size;
  const int V = GRU_V, E = GRU_E, H = GRU_H, B = GRU_B, T = GRU_T;
  const int M = B * T;           // 32768
  const int N3 = 3 * H;          // 3072

  const int*   x    = (const int*)d_in[0];
  const float* emb  = (const float*)d_in[1];
  const float* Wih0 = (const float*)d_in[2];
  const float* Whh0 = (const float*)d_in[3];
  const float* bih0 = (const float*)d_in[4];
  const float* bhh0 = (const float*)d_in[5];
  const float* Wih1 = (const float*)d_in[6];
  const float* Whh1 = (const float*)d_in[7];
  const float* bih1 = (const float*)d_in[8];
  const float* bhh1 = (const float*)d_in[9];
  const float* Wfc  = (const float*)d_in[10];
  const float* bfc  = (const float*)d_in[11];

  float* out = (float*)d_out;
  float* logits  = out;                        // [B,T,V]
  float* hfinal0 = out + (size_t)M * V;        // [B,H]
  float* hfinal1 = hfinal0 + (size_t)B * H;    // [B,H]

  // workspace carve-up (256B aligned)
  char* ws = (char*)d_ws;
  size_t off = 0;
  auto carve = [&](size_t bytes) { char* p = ws + off; off += (bytes + 255) & ~(size_t)255; return p; };
  __bf16* A0    = (__bf16*)carve((size_t)M * E * 2);
  __bf16* Wih0b = (__bf16*)carve((size_t)N3 * E * 2);
  __bf16* Whh0b = (__bf16*)carve((size_t)N3 * H * 2);
  __bf16* Wih1b = (__bf16*)carve((size_t)N3 * H * 2);
  __bf16* Whh1b = (__bf16*)carve((size_t)N3 * H * 2);
  __bf16* Wfcb  = (__bf16*)carve((size_t)V * H * 2);
  float*  gxbuf = (float*)carve((size_t)T * B * N3 * 4);   // reused for both layers
  __bf16* obuf  = (__bf16*)carve((size_t)M * H * 2);       // reused (o0 then o1)
  __bf16* hbuf  = (__bf16*)carve((size_t)2 * B * H * 2);   // double-buffered h
  unsigned* bar = (unsigned*)carve(256);

  auto cvt = [&](const float* s, __bf16* d, int n) {
    f32_to_bf16_kernel<<<(n + 255) / 256, 256, 0, stream>>>(s, d, n);
  };

  // 1) weights -> bf16, embedding gather -> bf16
  cvt(Wih0, Wih0b, N3 * E);
  cvt(Whh0, Whh0b, N3 * H);
  cvt(Wih1, Wih1b, N3 * H);
  cvt(Whh1, Whh1b, N3 * H);
  cvt(Wfc,  Wfcb,  V * H);
  {
    int total = M * E;
    gather_emb_kernel<<<(total + 255) / 256, 256, 0, stream>>>(x, emb, A0, total);
  }

  const int wavesPerBlk = 8, blkThreads = 256;
  const int scanBlocks  = 16;                      // H/64 column owners
  const size_t scanLds  = (size_t)B * H * 2 + (size_t)3 * B * 64 * 4; // 180224 B

  // 2) gx0 = emb[x] @ Wih0^T + bih0   -> gxbuf laid out [T,B,3H]
  {
    long waves = (long)(M / 16) * (N3 / 64);
    wmma_gemm_bias_kernel<<<(int)((waves + wavesPerBlk - 1) / wavesPerBlk), blkThreads, 0, stream>>>(
        A0, E, Wih0b, E, bih0, gxbuf, N3, M, N3, E, T, B, 1);
  }

  // 3) layer-0 recurrence
  {
    int n = 2 * B * H;
    zero_state_kernel<<<(n + 255) / 256, 256, 0, stream>>>(hbuf, n, bar);
    gru_scan_kernel<<<scanBlocks, 512, scanLds, stream>>>(Whh0b, bhh0, gxbuf, hbuf, obuf,
                                                          hfinal0, T, bar, scanBlocks);
  }

  // 4) gx1 = o0 @ Wih1^T + bih1 -> gxbuf [T,B,3H]
  {
    long waves = (long)(M / 16) * (N3 / 64);
    wmma_gemm_bias_kernel<<<(int)((waves + wavesPerBlk - 1) / wavesPerBlk), blkThreads, 0, stream>>>(
        obuf, H, Wih1b, H, bih1, gxbuf, N3, M, N3, H, T, B, 1);
  }

  // 5) layer-1 recurrence (overwrites obuf with o1)
  {
    int n = 2 * B * H;
    zero_state_kernel<<<(n + 255) / 256, 256, 0, stream>>>(hbuf, n, bar);
    gru_scan_kernel<<<scanBlocks, 512, scanLds, stream>>>(Whh1b, bhh1, gxbuf, hbuf, obuf,
                                                          hfinal1, T, bar, scanBlocks);
  }

  // 6) logits = o1 @ Wfc^T + bfc -> d_out [B,T,V]
  {
    long waves = (long)(M / 16) * (V / 64);
    wmma_gemm_bias_kernel<<<(int)((waves + wavesPerBlk - 1) / wavesPerBlk), blkThreads, 0, stream>>>(
        obuf, H, Wfcb, H, bfc, logits, V, M, V, H, T, B, 0);
  }
}
